// Rwkv7SelfAttention_57372173140344
// MI455X (gfx1250) — matrix operand order repaired and checked by reading, not compiled, not run on hardware
//
#include <hip/hip_runtime.h>

// RWKV7 block for MI455X (gfx1250): bf16 WMMA GEMMs (TDM-staged tiles) +
// register-resident sequential scan. Sizes fixed by the reference.
#define D_  2048
#define S_  2048
#define H_  32
#define N_  64
#define LW_ 64
#define LA_ 64
#define LG_ 128
#define LV_ 32
#define SD_ (S_*D_)

typedef __bf16 bf16_t;
typedef __attribute__((ext_vector_type(16))) __bf16 v16bf;
typedef __attribute__((ext_vector_type(8)))  __bf16 v8bf;
typedef __attribute__((ext_vector_type(8)))  float  v8f;
typedef unsigned int v4u __attribute__((ext_vector_type(4)));
typedef int          v8i __attribute__((ext_vector_type(8)));
typedef int          v4i __attribute__((ext_vector_type(4)));

#if defined(__has_builtin)
#if __has_builtin(__builtin_amdgcn_tensor_load_to_lds) && __has_builtin(__builtin_amdgcn_s_wait_tensorcnt)
#define HAVE_TDM 1
#endif
#endif

__device__ __forceinline__ float sigmoidf_(float x) { return 1.0f / (1.0f + __expf(-x)); }

__device__ __forceinline__ float block_reduce(float v, float* red, int nt) {
    int tid = threadIdx.x;
    red[tid] = v;
    __syncthreads();
    for (int off = nt >> 1; off > 0; off >>= 1) {
        if (tid < off) red[tid] += red[tid + off];
        __syncthreads();
    }
    float r = red[0];
    __syncthreads();
    return r;
}

// ---------------------------------------------------------------------------
// TDM: issue TENSOR_LOAD_TO_LDS of a 2D tile (tile_d0 x tile_d1, bf16) from a
// row-major tensor (tensor_d0 elems/row, tensor_d1 rows). LDS destination gets
// 8 DWORDs of pad after every 16 DWORDs (64B row) -> 96B padded row stride.
// Descriptor bitfields per CDNA5 ISA 8.3/8.4. OOB rows/cols read as zero.
// ---------------------------------------------------------------------------
#ifdef HAVE_TDM
__device__ __forceinline__ void tdm_load_tile_bf16(unsigned lds_off, const bf16_t* gptr,
                                                   unsigned tensor_d0, unsigned tensor_d1,
                                                   unsigned tile_d0, unsigned tile_d1)
{
    unsigned long long ga = (unsigned long long)(size_t)gptr;
    v4u g0;
    g0[0] = 1u;                                   // count=1, user descriptor
    g0[1] = lds_off;                              // lds_addr (bytes)
    g0[2] = (unsigned)ga;                         // global_addr[31:0]
    g0[3] = (unsigned)((ga >> 32) & 0x1FFFFFFu)   // global_addr[56:32]
          | (2u << 30);                           // type=2 ("image")
    v8i g1;
    g1[0] = (int)((1u << 16)                      // data_size = 2 bytes
                | (1u << 20)                      // pad_enable
                | (3u << 22)                      // pad_interval: 16 DWORDs
                | (7u << 25));                    // pad_amount:   8 DWORDs
    g1[1] = (int)((tensor_d0 & 0xFFFFu) << 16);                               // dim0[15:0]
    g1[2] = (int)(((tensor_d0 >> 16) & 0xFFFFu) | ((tensor_d1 & 0xFFFFu) << 16));
    g1[3] = (int)(((tensor_d1 >> 16) & 0xFFFFu) | ((tile_d0 & 0xFFFFu) << 16));
    g1[4] = (int)(tile_d1 & 0xFFFFu);                                         // tile_dim1
    g1[5] = (int)tensor_d0;                       // tensor_dim0_stride[31:0] = K
    g1[6] = 0;                                    // stride[47:32] | dim1_stride lo
    g1[7] = 0;
    v4i z4 = {0, 0, 0, 0};
#if defined(__clang_major__) && (__clang_major__ >= 23)
    v8i z8 = {0, 0, 0, 0, 0, 0, 0, 0};
    __builtin_amdgcn_tensor_load_to_lds(g0, g1, z4, z4, z8, 0);
#else
    __builtin_amdgcn_tensor_load_to_lds(g0, g1, z4, z4, 0);
#endif
}
#endif

// ---------------------------------------------------------------------------
// bf16 WMMA GEMM: C[M x Ncols] = A[M x K] * Bt[Ncols x K]^T (both K-contiguous)
// Block tile 128x128x32, 8 waves; each wave: 32 rows x 64 cols =
// 2 A-frags x 4 B-frags -> 8 v_wmma per K-step. Tiles staged via TDM
// (double-buffered, TENSORcnt) or plain loads as fallback.
// ---------------------------------------------------------------------------
#define BM 128
#define BN 128
#define BK 32
#define LP 48   // padded LDS row stride in bf16 elems (96B, matches TDM pad)

__global__ __launch_bounds__(256)
void gemm_bf16_wmma(const bf16_t* __restrict__ A, const bf16_t* __restrict__ Bt,
                    float* __restrict__ C, int M, int Ncols, int K)
{
    __shared__ bf16_t As[2][BM][LP];
    __shared__ bf16_t Bs[2][BN][LP];
    const int tid  = threadIdx.x;
    const int lane = tid & 31;
    const int wave = tid >> 5;
    const int bm = blockIdx.y * BM;
    const int bn = blockIdx.x * BN;
    const int m0 = (wave & 3) * 32;   // wave's 32-row strip
    const int n0 = (wave >> 2) * 64;  // wave's 64-col strip

    v8f acc[2][4] = {};

    // Fragment addressing per CDNA5 ISA 16-bit layouts:
    // A (16x32): lanes 0-15 row=lane, K{0..7,16..23}; lanes 16-31 K{8..15,24..31}
    // B (32x16): lanes 0-15 col=lane, K{0..7,16..23}; lanes 16-31 K{8..15,24..31}
    const int l15   = lane & 15;
    const int akoff = (lane < 16) ? 0 : 8;
    const int bkoff = (lane < 16) ? 0 : 8;
    const int nsteps = K / BK;

#ifdef HAVE_TDM
    if (wave == 0) {  // prologue: DMA tiles for step 0 (descriptors are uniform)
        tdm_load_tile_bf16((unsigned)(size_t)&As[0][0][0], A + (size_t)bm * K,
                           (unsigned)K, (unsigned)M, BK, BM);
        tdm_load_tile_bf16((unsigned)(size_t)&Bs[0][0][0], Bt + (size_t)bn * K,
                           (unsigned)K, (unsigned)Ncols, BK, BN);
    }
#endif
    for (int s = 0; s < nsteps; ++s) {
        const int buf = s & 1;
#ifdef HAVE_TDM
        if (wave == 0) {
            if (s + 1 < nsteps) {
                const int nb = (s + 1) & 1;
                const int k1 = (s + 1) * BK;
                tdm_load_tile_bf16((unsigned)(size_t)&As[nb][0][0],
                                   A + (size_t)bm * K + k1, (unsigned)K, (unsigned)M, BK, BM);
                tdm_load_tile_bf16((unsigned)(size_t)&Bs[nb][0][0],
                                   Bt + (size_t)bn * K + k1, (unsigned)K, (unsigned)Ncols, BK, BN);
                __builtin_amdgcn_s_wait_tensorcnt(2);  // step-s tiles complete
            } else {
                __builtin_amdgcn_s_wait_tensorcnt(0);
            }
        }
        __syncthreads();
#else
        {   // fallback: cooperative loads, 16 bf16 per thread per tile
            const int lr = tid >> 1;
            const int lc = (tid & 1) * 16;
            const int k0 = s * BK;
            v8bf a0 = *(const v8bf*)(A + (size_t)(bm + lr) * K + k0 + lc);
            v8bf a1 = *(const v8bf*)(A + (size_t)(bm + lr) * K + k0 + lc + 8);
            v8bf b0 = {}, b1 = {};
            if (bn + lr < Ncols) {
                b0 = *(const v8bf*)(Bt + (size_t)(bn + lr) * K + k0 + lc);
                b1 = *(const v8bf*)(Bt + (size_t)(bn + lr) * K + k0 + lc + 8);
            }
            *(v8bf*)&As[buf][lr][lc]     = a0;
            *(v8bf*)&As[buf][lr][lc + 8] = a1;
            *(v8bf*)&Bs[buf][lr][lc]     = b0;
            *(v8bf*)&Bs[buf][lr][lc + 8] = b1;
        }
        __syncthreads();
#endif
        union frag { v16bf v; v8bf h[2]; };
        frag af[2], bfr[4];
        #pragma unroll
        for (int ra = 0; ra < 2; ++ra) {
            af[ra].h[0] = *(const v8bf*)&As[buf][m0 + ra * 16 + l15][akoff];
            af[ra].h[1] = *(const v8bf*)&As[buf][m0 + ra * 16 + l15][akoff + 16];
        }
        #pragma unroll
        for (int c = 0; c < 4; ++c) {
            bfr[c].h[0] = *(const v8bf*)&Bs[buf][n0 + c * 16 + l15][bkoff];
            bfr[c].h[1] = *(const v8bf*)&Bs[buf][n0 + c * 16 + l15][bkoff + 16];
        }
        #pragma unroll
        for (int ra = 0; ra < 2; ++ra)
            #pragma unroll
            for (int c = 0; c < 4; ++c)
                acc[ra][c] = __builtin_amdgcn_wmma_f32_16x16x32_bf16(
                    false, af[ra].v, false, bfr[c].v, (short)0, acc[ra][c], false, false);
        __syncthreads();
    }

    // C/D layout: VGPR r -> M = r (lanes<16) or 8+r (lanes>=16); N = lane&15
    const int rsel = (lane < 16) ? 0 : 8;
    #pragma unroll
    for (int ra = 0; ra < 2; ++ra) {
        const int crow = bm + m0 + ra * 16 + rsel;
        #pragma unroll
        for (int c = 0; c < 4; ++c) {
            const int ccol = bn + n0 + c * 16 + l15;
            if (ccol < Ncols) {
                #pragma unroll
                for (int r2 = 0; r2 < 8; ++r2)
                    C[(size_t)(crow + r2) * Ncols + ccol] = acc[ra][c][r2];
            }
        }
    }
}

// ---------------------------------------------------------------------------
// LayerNorm + token-shift mixing. One block per token; recompute previous
// row's LN instead of materializing xn (pass is bandwidth-bound).
// ---------------------------------------------------------------------------
__global__ __launch_bounds__(256)
void ln_mix_kernel(const float* __restrict__ x, const float* __restrict__ state1,
                   const float* __restrict__ lnw, const float* __restrict__ lnb,
                   const float* __restrict__ mr, const float* __restrict__ mw,
                   const float* __restrict__ mk, const float* __restrict__ mv,
                   const float* __restrict__ ma, const float* __restrict__ mg,
                   bf16_t* __restrict__ xr, bf16_t* __restrict__ xw,
                   bf16_t* __restrict__ xk, bf16_t* __restrict__ xv,
                   bf16_t* __restrict__ xa, bf16_t* __restrict__ xg,
                   float* __restrict__ state1_out)
{
    __shared__ float red[256];
    const int s   = blockIdx.x;
    const int tid = threadIdx.x;
    float cur[8], prv[8];
    #pragma unroll
    for (int q = 0; q < 8; ++q) {
        int d  = tid + q * 256;
        cur[q] = x[(size_t)s * D_ + d];
        prv[q] = (s == 0) ? state1[d] : x[(size_t)(s - 1) * D_ + d];
    }
    float sc = 0.f, sc2 = 0.f, sp = 0.f, sp2 = 0.f;
    #pragma unroll
    for (int q = 0; q < 8; ++q) {
        sc += cur[q]; sc2 += cur[q] * cur[q];
        sp += prv[q]; sp2 += prv[q] * prv[q];
    }
    sc  = block_reduce(sc,  red, 256);
    sc2 = block_reduce(sc2, red, 256);
    sp  = block_reduce(sp,  red, 256);
    sp2 = block_reduce(sp2, red, 256);

    const float mc = sc * (1.0f / D_);
    const float rc = rsqrtf(sc2 * (1.0f / D_) - mc * mc + 1e-5f);
    float mp = 0.f, rp = 0.f;
    if (s > 0) {
        mp = sp * (1.0f / D_);
        rp = rsqrtf(sp2 * (1.0f / D_) - mp * mp + 1e-5f);
    }
    #pragma unroll
    for (int q = 0; q < 8; ++q) {
        int d = tid + q * 256;
        float xn = (cur[q] - mc) * rc * lnw[d] + lnb[d];
        float xp = (s == 0) ? prv[q] : (prv[q] - mp) * rp * lnw[d] + lnb[d];
        float sx = xp - xn;
        size_t o = (size_t)s * D_ + d;
        xr[o] = (bf16_t)(xn + sx * mr[d]);
        xw[o] = (bf16_t)(xn + sx * mw[d]);
        xk[o] = (bf16_t)(xn + sx * mk[d]);
        xv[o] = (bf16_t)(xn + sx * mv[d]);
        xa[o] = (bf16_t)(xn + sx * ma[d]);
        xg[o] = (bf16_t)(xn + sx * mg[d]);
        if (s == S_ - 1) state1_out[d] = xn;
    }
}

// ---------------------------------------------------------------------------
// Small helpers: f32->bf16, f32 transpose->bf16, activation->bf16
// ---------------------------------------------------------------------------
__global__ void f32_to_bf16(const float* __restrict__ in, bf16_t* __restrict__ out, int n) {
    int i = blockIdx.x * 256 + threadIdx.x;
    if (i < n) out[i] = (bf16_t)in[i];
}

__global__ void transpose_to_bf16(const float* __restrict__ in, bf16_t* __restrict__ out,
                                  int rows, int cols) {  // in[r][c] -> out[c][r]
    int i = blockIdx.x * 256 + threadIdx.x;
    if (i < rows * cols) {
        int r = i / cols, c = i % cols;
        out[(size_t)c * rows + r] = (bf16_t)in[i];
    }
}

__global__ void act_to_bf16(const float* __restrict__ in, bf16_t* __restrict__ out,
                            int n, int mode) {  // 0=tanh, 1=sigmoid, 2=identity
    int i = blockIdx.x * 256 + threadIdx.x;
    if (i < n) {
        float v = in[i];
        if (mode == 0) v = tanhf(v);
        else if (mode == 1) v = sigmoidf_(v);
        out[i] = (bf16_t)v;
    }
}

// ---------------------------------------------------------------------------
// Per-(s,h) preprocessing; everything in-place over raw GEMM outputs:
//   kbuf -> k*(1+(a-1)*k_a), vbuf -> v + (vf-v)*sig(...), tdbuf -> w_dec,
//   abuf -> b = kk_hat*a, vmixbuf -> a_neg = -kk_hat
// ---------------------------------------------------------------------------
__global__ __launch_bounds__(64)
void prep_kernel(float* __restrict__ kbuf, float* __restrict__ vbuf,
                 float* __restrict__ tdbuf, float* __restrict__ abuf,
                 float* __restrict__ vmixbuf,
                 const float* __restrict__ v_first,
                 const float* __restrict__ k_k, const float* __restrict__ k_a,
                 const float* __restrict__ a0, const float* __restrict__ w0,
                 const float* __restrict__ v0)
{
    __shared__ float red[64];
    const int idx = blockIdx.x;          // s*H + h
    const int j   = threadIdx.x;
    const int h   = idx % H_;
    const int d   = h * N_ + j;
    const size_t base = (size_t)idx * N_ + j;

    float kraw = kbuf[base];
    float av   = sigmoidf_(abuf[base] + a0[d]);
    float kk   = kraw * k_k[d];
    float ss   = block_reduce(kk * kk, red, 64);
    float dn   = fmaxf(sqrtf(ss), 1e-12f);
    kk /= dn;

    float kmod = kraw * (1.0f + (av - 1.0f) * k_a[d]);
    float vraw = vbuf[base];
    float vm   = sigmoidf_(vmixbuf[base] + v0[d]);
    float vmod = vraw + (v_first[base] - vraw) * vm;
    float wdec = __expf(-0.606531f * sigmoidf_(tdbuf[base] + w0[d]));

    kbuf[base]    = kmod;
    vbuf[base]    = vmod;
    tdbuf[base]   = wdec;
    abuf[base]    = kk * av;   // b
    vmixbuf[base] = -kk;       // a_neg
}

// ---------------------------------------------------------------------------
// Sequential WKV scan. One block per head; thread i keeps state row i (64
// floats) in VGPRs; per-step j-vectors broadcast via LDS.
// ---------------------------------------------------------------------------
__global__ __launch_bounds__(64)
void scan_kernel(const float* __restrict__ rbuf, const float* __restrict__ wbuf,
                 const float* __restrict__ kbuf, const float* __restrict__ vbuf,
                 const float* __restrict__ anbuf, const float* __restrict__ bbuf,
                 const float* __restrict__ state2_in,
                 float* __restrict__ ybuf, float* __restrict__ state2_out)
{
    const int h = blockIdx.x;
    const int i = threadIdx.x;
    float srow[N_];
    #pragma unroll
    for (int j = 0; j < N_; ++j)
        srow[j] = state2_in[(size_t)h * N_ * N_ + i * N_ + j];

    __shared__ float wv[N_], kv[N_], avv[N_], bvv[N_], rv[N_];
    for (int t = 0; t < S_; ++t) {
        const size_t rb = (size_t)t * D_ + h * N_;
        wv[i]  = wbuf[rb + i];
        kv[i]  = kbuf[rb + i];
        avv[i] = anbuf[rb + i];
        bvv[i] = bbuf[rb + i];
        rv[i]  = rbuf[rb + i];
        const float vi = vbuf[rb + i];
        __syncthreads();
        float sa = 0.f;
        #pragma unroll
        for (int j = 0; j < N_; ++j) sa += srow[j] * avv[j];
        float y = 0.f;
        #pragma unroll
        for (int j = 0; j < N_; ++j) {
            srow[j] = srow[j] * wv[j] + sa * bvv[j] + vi * kv[j];
            y += srow[j] * rv[j];
        }
        ybuf[rb + i] = y;
        __syncthreads();
    }
    #pragma unroll
    for (int j = 0; j < N_; ++j)
        state2_out[(size_t)h * N_ * N_ + i * N_ + j] = srow[j];
}

// ---------------------------------------------------------------------------
// Per-(s,h) group-norm of y, +rkv, *gate -> bf16 for the final GEMM.
// ---------------------------------------------------------------------------
__global__ __launch_bounds__(64)
void out_kernel(const float* __restrict__ ybuf, const float* __restrict__ rbuf,
                const float* __restrict__ kbuf, const float* __restrict__ vbuf,
                const float* __restrict__ gate, const float* __restrict__ r_k,
                const float* __restrict__ lnxw, const float* __restrict__ lnxb,
                bf16_t* __restrict__ obuf)
{
    __shared__ float red[64];
    const int idx = blockIdx.x;
    const int j   = threadIdx.x;
    const int h   = idx % H_;
    const int d   = h * N_ + j;
    const size_t base = (size_t)idx * N_ + j;

    float yv = ybuf[base];
    float m  = block_reduce(yv, red, 64) * (1.0f / N_);
    float vv = block_reduce(yv * yv, red, 64) * (1.0f / N_) - m * m;
    float o  = (yv - m) * rsqrtf(vv + 0.00064f);
    o = o * lnxw[d] + lnxb[d];

    float tdot = block_reduce(rbuf[base] * kbuf[base] * r_k[d], red, 64);
    float rkv  = tdot * vbuf[base];
    obuf[base] = (bf16_t)((o + rkv) * gate[base]);
}

__global__ void residual_add(const float* __restrict__ x, const float* __restrict__ osum,
                             float* __restrict__ out, int n) {
    int i = blockIdx.x * 256 + threadIdx.x;
    if (i < n) out[i] = x[i] + osum[i];
}

// ---------------------------------------------------------------------------
extern "C" void kernel_launch(void* const* d_in, const int* in_sizes, int n_in,
                              void* d_out, int out_size, void* d_ws, size_t ws_size,
                              hipStream_t stream)
{
    const float* x       = (const float*)d_in[0];
    const float* state1  = (const float*)d_in[1];
    const float* state2  = (const float*)d_in[2];
    const float* v_first = (const float*)d_in[3];
    const float* ln1w    = (const float*)d_in[4];
    const float* ln1b    = (const float*)d_in[5];
    const float* x_r     = (const float*)d_in[6];
    const float* x_w     = (const float*)d_in[7];
    const float* x_k     = (const float*)d_in[8];
    const float* x_v     = (const float*)d_in[9];
    const float* x_a     = (const float*)d_in[10];
    const float* x_g     = (const float*)d_in[11];
    const float* Wr      = (const float*)d_in[12];
    const float* Wk      = (const float*)d_in[13];
    const float* Wv      = (const float*)d_in[14];
    const float* Wo      = (const float*)d_in[15];
    const float* w1      = (const float*)d_in[16];
    const float* w2      = (const float*)d_in[17];
    const float* w0      = (const float*)d_in[18];
    const float* a1      = (const float*)d_in[19];
    const float* a2      = (const float*)d_in[20];
    const float* a0      = (const float*)d_in[21];
    const float* g1      = (const float*)d_in[22];
    const float* g2      = (const float*)d_in[23];
    const float* v1      = (const float*)d_in[24];
    const float* v2      = (const float*)d_in[25];
    const float* v0      = (const float*)d_in[26];
    const float* k_k     = (const float*)d_in[27];
    const float* k_a     = (const float*)d_in[28];
    const float* r_k     = (const float*)d_in[29];
    const float* ln_x_w  = (const float*)d_in[30];
    const float* ln_x_b  = (const float*)d_in[31];

    float* out        = (float*)d_out;
    float* state1_out = out + SD_;
    float* state2_out = out + SD_ + D_;

    char* p = (char*)d_ws;
    auto carve = [&](size_t bytes) -> void* {
        void* r = (void*)p;
        p += (bytes + 255) & ~(size_t)255;
        return r;
    };

    bf16_t* xr_bf = (bf16_t*)carve((size_t)SD_ * 2);
    bf16_t* xw_bf = (bf16_t*)carve((size_t)SD_ * 2);
    bf16_t* xk_bf = (bf16_t*)carve((size_t)SD_ * 2);
    bf16_t* xv_bf = (bf16_t*)carve((size_t)SD_ * 2);
    bf16_t* xa_bf = (bf16_t*)carve((size_t)SD_ * 2);
    bf16_t* xg_bf = (bf16_t*)carve((size_t)SD_ * 2);
    bf16_t* Wr_bf = (bf16_t*)carve((size_t)SD_ * 2);
    bf16_t* Wk_bf = (bf16_t*)carve((size_t)SD_ * 2);
    bf16_t* Wv_bf = (bf16_t*)carve((size_t)SD_ * 2);
    bf16_t* Wo_bf = (bf16_t*)carve((size_t)SD_ * 2);
    bf16_t* w1t = (bf16_t*)carve((size_t)D_ * LW_ * 2);
    bf16_t* w2t = (bf16_t*)carve((size_t)D_ * LW_ * 2);
    bf16_t* a1t = (bf16_t*)carve((size_t)D_ * LA_ * 2);
    bf16_t* a2t = (bf16_t*)carve((size_t)D_ * LA_ * 2);
    bf16_t* g1t = (bf16_t*)carve((size_t)D_ * LG_ * 2);
    bf16_t* g2t = (bf16_t*)carve((size_t)D_ * LG_ * 2);
    bf16_t* v1t = (bf16_t*)carve((size_t)D_ * LV_ * 2);
    bf16_t* v2t = (bf16_t*)carve((size_t)D_ * LV_ * 2);

    float* rbuf    = (float*)carve((size_t)SD_ * 4);
    float* kbuf    = (float*)carve((size_t)SD_ * 4);
    float* vbuf    = (float*)carve((size_t)SD_ * 4);
    float* tdbuf   = (float*)carve((size_t)SD_ * 4);  // w_dec, later reused as osum
    float* abuf    = (float*)carve((size_t)SD_ * 4);  // a_raw -> b
    float* vmixbuf = (float*)carve((size_t)SD_ * 4);  // vmix_raw -> a_neg
    float* gatebuf = (float*)carve((size_t)SD_ * 4);
    float* ybuf    = (float*)carve((size_t)SD_ * 4);

    float*  hw    = (float*) carve((size_t)S_ * LW_ * 4);
    bf16_t* hw_bf = (bf16_t*)carve((size_t)S_ * LW_ * 2);
    float*  ha    = (float*) carve((size_t)S_ * LA_ * 4);
    bf16_t* ha_bf = (bf16_t*)carve((size_t)S_ * LA_ * 2);
    float*  hg    = (float*) carve((size_t)S_ * LG_ * 4);
    bf16_t* hg_bf = (bf16_t*)carve((size_t)S_ * LG_ * 2);
    float*  hv    = (float*) carve((size_t)S_ * LV_ * 4);
    bf16_t* hv_bf = (bf16_t*)carve((size_t)S_ * LV_ * 2);
    bf16_t* obuf  = (bf16_t*)carve((size_t)SD_ * 2);
    float*  osum  = tdbuf;  // reuse: w_dec consumed before final GEMM

    const int g1d = (SD_ + 255) / 256;

    // Weight precision conversion (bf16 for WMMA); W? already [Ncols x K].
    f32_to_bf16<<<g1d, 256, 0, stream>>>(Wr, Wr_bf, SD_);
    f32_to_bf16<<<g1d, 256, 0, stream>>>(Wk, Wk_bf, SD_);
    f32_to_bf16<<<g1d, 256, 0, stream>>>(Wv, Wv_bf, SD_);
    f32_to_bf16<<<g1d, 256, 0, stream>>>(Wo, Wo_bf, SD_);
    transpose_to_bf16<<<(D_*LW_ + 255)/256, 256, 0, stream>>>(w1, w1t, D_, LW_);
    transpose_to_bf16<<<(LW_*D_ + 255)/256, 256, 0, stream>>>(w2, w2t, LW_, D_);
    transpose_to_bf16<<<(D_*LA_ + 255)/256, 256, 0, stream>>>(a1, a1t, D_, LA_);
    transpose_to_bf16<<<(LA_*D_ + 255)/256, 256, 0, stream>>>(a2, a2t, LA_, D_);
    transpose_to_bf16<<<(D_*LG_ + 255)/256, 256, 0, stream>>>(g1, g1t, D_, LG_);
    transpose_to_bf16<<<(LG_*D_ + 255)/256, 256, 0, stream>>>(g2, g2t, LG_, D_);
    transpose_to_bf16<<<(D_*LV_ + 255)/256, 256, 0, stream>>>(v1, v1t, D_, LV_);
    transpose_to_bf16<<<(LV_*D_ + 255)/256, 256, 0, stream>>>(v2, v2t, LV_, D_);

    // LN + token-shift mixes (also writes state1_out).
    ln_mix_kernel<<<S_, 256, 0, stream>>>(x, state1, ln1w, ln1b,
                                          x_r, x_w, x_k, x_v, x_a, x_g,
                                          xr_bf, xw_bf, xk_bf, xv_bf, xa_bf, xg_bf,
                                          state1_out);

    dim3 gBig(D_ / BN, S_ / BM);
    dim3 gNar(1, S_ / BM);  // all LoRA down-projections have Ncols <= BN
    gemm_bf16_wmma<<<gBig, 256, 0, stream>>>(xr_bf, Wr_bf, rbuf, S_, D_, D_);
    gemm_bf16_wmma<<<gBig, 256, 0, stream>>>(xk_bf, Wk_bf, kbuf, S_, D_, D_);
    gemm_bf16_wmma<<<gBig, 256, 0, stream>>>(xv_bf, Wv_bf, vbuf, S_, D_, D_);

    // LoRA projections.
    gemm_bf16_wmma<<<gNar, 256, 0, stream>>>(xw_bf, w1t, hw, S_, LW_, D_);
    act_to_bf16<<<(S_*LW_ + 255)/256, 256, 0, stream>>>(hw, hw_bf, S_*LW_, 0);
    gemm_bf16_wmma<<<gBig, 256, 0, stream>>>(hw_bf, w2t, tdbuf, S_, D_, LW_);

    gemm_bf16_wmma<<<gNar, 256, 0, stream>>>(xa_bf, a1t, ha, S_, LA_, D_);
    act_to_bf16<<<(S_*LA_ + 255)/256, 256, 0, stream>>>(ha, ha_bf, S_*LA_, 2);
    gemm_bf16_wmma<<<gBig, 256, 0, stream>>>(ha_bf, a2t, abuf, S_, D_, LA_);

    gemm_bf16_wmma<<<gNar, 256, 0, stream>>>(xg_bf, g1t, hg, S_, LG_, D_);
    act_to_bf16<<<(S_*LG_ + 255)/256, 256, 0, stream>>>(hg, hg_bf, S_*LG_, 1);
    gemm_bf16_wmma<<<gBig, 256, 0, stream>>>(hg_bf, g2t, gatebuf, S_, D_, LG_);

    gemm_bf16_wmma<<<gNar, 256, 0, stream>>>(xv_bf, v1t, hv, S_, LV_, D_);
    act_to_bf16<<<(S_*LV_ + 255)/256, 256, 0, stream>>>(hv, hv_bf, S_*LV_, 2);
    gemm_bf16_wmma<<<gBig, 256, 0, stream>>>(hv_bf, v2t, vmixbuf, S_, D_, LV_);

    // Per-token per-head transforms (in-place).
    prep_kernel<<<S_ * H_, 64, 0, stream>>>(kbuf, vbuf, tdbuf, abuf, vmixbuf,
                                            v_first, k_k, k_a, a0, w0, v0);

    // Sequential WKV scan; writes state2_out.
    scan_kernel<<<H_, 64, 0, stream>>>(rbuf, tdbuf, kbuf, vbuf, vmixbuf, abuf,
                                       state2, ybuf, state2_out);

    // Group-norm + rkv + gate -> bf16.
    out_kernel<<<S_ * H_, 64, 0, stream>>>(ybuf, rbuf, kbuf, vbuf, gatebuf,
                                           r_k, ln_x_w, ln_x_b, obuf);

    // Final projection + residual.
    gemm_bf16_wmma<<<gBig, 256, 0, stream>>>(obuf, Wo_bf, osum, S_, D_, D_);
    residual_add<<<g1d, 256, 0, stream>>>(x, osum, out, SD_);
}
